// GraphAttentionLayer_21912923144629
// MI455X (gfx1250) — matrix operand order repaired and checked by reading, not compiled, not run on hardware
//
#include <hip/hip_runtime.h>
#include <math.h>

typedef __bf16 bf16_t;
typedef __attribute__((ext_vector_type(16))) __bf16 v16bf;
typedef __attribute__((ext_vector_type(8)))  __bf16 v8bf;
typedef __attribute__((ext_vector_type(8)))  float  v8f;
typedef __attribute__((ext_vector_type(4)))  float  v4f;

#define BATCH 8
#define NTOK  2048
#define DF    128
#define NEG_INF_REF (-9e15f)
#define PSTRIDE 72   // P-tile row stride in elements (64 cols + 8 pad, 144B, 16B-aligned)

__device__ __forceinline__ v8f zero8() {
    v8f z;
#pragma unroll
    for (int i = 0; i < 8; i++) z[i] = 0.f;
    return z;
}

// A-matrix 16x32 bf16 fragment (contiguous K per lane-half): p points at row start + c0
// elements [0..7] at p[0..7], elements [8..15] at p[8..15]
__device__ __forceinline__ v16bf load_frag_A(const bf16_t* p) {
    v8bf a = *(const v8bf*)p;
    v8bf b = *(const v8bf*)(p + 8);
    v16bf r;
#pragma unroll
    for (int i = 0; i < 8; i++) { r[i] = a[i]; r[8 + i] = b[i]; }
    return r;
}

// B-matrix 32x16 bf16 fragment (contiguous K per lane-half): p points at 16
// contiguous elements (K = 16*half + 0..15 of column lq)
__device__ __forceinline__ v16bf load_frag_B(const bf16_t* p) {
    v8bf a = *(const v8bf*)p;
    v8bf b = *(const v8bf*)(p + 8);
    v16bf r;
#pragma unroll
    for (int i = 0; i < 8; i++) { r[i] = a[i]; r[8 + i] = b[i]; }
    return r;
}

#define WMMA_BF16(A, B, C) \
    __builtin_amdgcn_wmma_f32_16x16x32_bf16(false, (A), false, (B), (short)0, (C), false, false)

// ---------------------------------------------------------------------------
// Kernel 1: xW = x @ W  (f32 in, bf16 out, f32 accumulate)
// Writes xW row-major [b][n][d] and xWT [b][d][n] (both bf16) to workspace.
// Block = 128 threads (4 waves), each wave owns one 16-row tile.
// ---------------------------------------------------------------------------
__global__ __launch_bounds__(128) void gat_xw_kernel(
    const float* __restrict__ x, const float* __restrict__ W,
    bf16_t* __restrict__ xW, bf16_t* __restrict__ xWT)
{
    __shared__ __align__(16) bf16_t WT[DF * 136];        // W transposed, padded stride
    __shared__ __align__(16) bf16_t stage[4 * DF * 16];  // per-wave [feature][m]

    const int tid  = threadIdx.x;
    const int lane = tid & 31;
    const int wid  = tid >> 5;
    const int lq   = lane & 15;
    const int h    = lane >> 4;
    const int c0   = h * 16;

    // transpose W into LDS as bf16: WT[n][k] = W[k][n]
#pragma unroll 4
    for (int k = 0; k < DF; k++)
        WT[tid * 136 + k] = (bf16_t)W[k * DF + tid];
    __syncthreads();

    const int tile = blockIdx.x * 4 + wid;
    const int b    = tile >> 7;          // 128 tiles per batch
    const int row0 = (tile & 127) << 4;

    // A fragments from x (convert f32 -> bf16), 4 K-chunks of 32
    v16bf Aq[4];
    const float* xrow = x + (size_t)(b * NTOK + row0 + lq) * DF;
#pragma unroll
    for (int c = 0; c < 4; c++) {
        const float* p = xrow + c * 32 + c0;
        v4f f0 = *(const v4f*)(p);
        v4f f1 = *(const v4f*)(p + 4);
        v4f f2 = *(const v4f*)(p + 8);
        v4f f3 = *(const v4f*)(p + 12);
        v16bf a;
#pragma unroll
        for (int i = 0; i < 4; i++) {
            a[i] = (bf16_t)f0[i];  a[4 + i]  = (bf16_t)f1[i];
            a[8 + i] = (bf16_t)f2[i]; a[12 + i] = (bf16_t)f3[i];
        }
        Aq[c] = a;
    }

    bf16_t* st = stage + wid * DF * 16;  // [feature][m], m = row within tile

#pragma unroll
    for (int n0 = 0; n0 < 8; n0++) {
        v8f D = zero8();
#pragma unroll
        for (int c = 0; c < 4; c++) {
            // B[k][n] = W[k][n0*16+n]: lane column lq, K = c*32 + 16h + 0..15
            const bf16_t* wp = WT + (n0 * 16 + lq) * 136 + c * 32 + h * 16;
            v16bf Bw = load_frag_B(wp);
            D = WMMA_BF16(Aq[c], Bw, D);
        }
#pragma unroll
        for (int r = 0; r < 8; r++) {
            const int m = r + 8 * h;
            bf16_t v = (bf16_t)D[r];
            xW[(size_t)(b * NTOK + row0 + m) * DF + n0 * 16 + lq] = v;
            st[(n0 * 16 + lq) * 16 + m] = v;  // transposed stage
        }
    }

    // vectorized xWT writes: each lane handles 4 feature rows, 32B each
#pragma unroll
    for (int i = 0; i < 4; i++) {
        const int f = lane + 32 * i;
        const v8bf* sp = (const v8bf*)(st + f * 16);
        bf16_t* dp = xWT + (size_t)(b * DF + f) * NTOK + row0;
        *(v8bf*)(dp)     = sp[0];
        *(v8bf*)(dp + 8) = sp[1];
    }
}

// ---------------------------------------------------------------------------
// Kernel 2: flash-attention-style GAT. One wave = one 16-row tile.
// 64-column inner tile amortizes softmax shuffles / O-rescale over 2x columns.
// Streams adj exactly once; xW/xWT are L2-resident bf16.
// ---------------------------------------------------------------------------
__global__ __launch_bounds__(128) void gat_attn_kernel(
    const float* __restrict__ adj,
    const bf16_t* __restrict__ xW, const bf16_t* __restrict__ xWT,
    float* __restrict__ out)
{
    __shared__ __align__(16) bf16_t Pt[4 * 16 * PSTRIDE];  // per-wave 16x64 P tile

    const int tid  = threadIdx.x;
    const int lane = tid & 31;
    const int wid  = tid >> 5;
    const int lq   = lane & 15;
    const int h    = lane >> 4;
    const int c0   = h * 16;

    const int tile = blockIdx.x * 4 + wid;
    const int b    = tile >> 7;
    const int row0 = (tile & 127) << 4;

    const bf16_t* xWb  = xW  + (size_t)b * NTOK * DF;
    const bf16_t* xWTb = xWT + (size_t)b * DF * NTOK;
    const float*  adjb = adj + (size_t)b * NTOK * NTOK;

    // Q fragments (A-layout), fixed for whole tile
    v16bf Aq[4];
    const bf16_t* qrow = xWb + (size_t)(row0 + lq) * DF;
#pragma unroll
    for (int c = 0; c < 4; c++)
        Aq[c] = load_frag_A(qrow + c * 32 + c0);

    float mrow[8], lrow[8];
    v8f O[8];
#pragma unroll
    for (int r = 0; r < 8; r++) { mrow[r] = -INFINITY; lrow[r] = 0.f; O[r] = zero8(); }

    bf16_t* pt = Pt + wid * 16 * PSTRIDE;

    for (int j0 = 0; j0 < NTOK; j0 += 64) {
        // ---- scores: S = Q @ (xW rows j0..j0+63)^T, four 16-col subtiles ----
        v8f S[4];
#pragma unroll
        for (int s = 0; s < 4; s++) S[s] = zero8();
#pragma unroll
        for (int c = 0; c < 4; c++) {
#pragma unroll
            for (int s = 0; s < 4; s++) {
                // B[k][n] = xW[j0+s*16+n][kbase+k]: lane col lq, contiguous K half
                const bf16_t* p = xWb + (size_t)(j0 + s * 16 + lq) * DF + c * 32 + h * 16;
                S[s] = WMMA_BF16(Aq[c], load_frag_B(p), S[s]);
            }
        }

        // ---- adj bias + mask + online softmax (per row r+8h) ----
        float alph[8];
#pragma unroll
        for (int r = 0; r < 8; r++) {
            const int m = r + 8 * h;
            const float* arow = adjb + (size_t)(row0 + m) * NTOK + j0;
            float sv[4];
#pragma unroll
            for (int s = 0; s < 4; s++) {
                float a = arow[s * 16 + lq];
                sv[s] = (a > 0.f) ? (S[s][r] + a) : NEG_INF_REF;
            }
            float cand = fmaxf(fmaxf(sv[0], sv[1]), fmaxf(sv[2], sv[3]));
#pragma unroll
            for (int msk = 1; msk < 16; msk <<= 1)
                cand = fmaxf(cand, __shfl_xor(cand, msk));
            float mnew  = fmaxf(mrow[r], cand);
            float alpha = __expf(mrow[r] - mnew);
            float pv[4], rs = 0.f;
#pragma unroll
            for (int s = 0; s < 4; s++) { pv[s] = __expf(sv[s] - mnew); rs += pv[s]; }
#pragma unroll
            for (int msk = 1; msk < 16; msk <<= 1)
                rs += __shfl_xor(rs, msk);
            lrow[r] = lrow[r] * alpha + rs;
            mrow[r] = mnew;
            alph[r] = alpha;

            // P tile to LDS (D-layout -> row-major 16x64)
#pragma unroll
            for (int s = 0; s < 4; s++)
                pt[m * PSTRIDE + s * 16 + lq] = (bf16_t)pv[s];
        }

        // rescale O accumulators by alpha (per row)
#pragma unroll
        for (int n0 = 0; n0 < 8; n0++)
#pragma unroll
            for (int r = 0; r < 8; r++)
                O[n0][r] *= alph[r];

        // ---- O += P @ V  (V = xW[j0..j0+63][:], B-frags from xWT) ----
        // two A-frags of P (K-chunks of 32); same-wave LDS ops are in-order
        v16bf Pf0 = load_frag_A(pt + lq * PSTRIDE + c0);
        v16bf Pf1 = load_frag_A(pt + lq * PSTRIDE + 32 + c0);
#pragma unroll
        for (int n0 = 0; n0 < 8; n0++) {
            // B[k][n] = xW[j0+k][n0*16+n] = xWT[n0*16+n][j0+k]
            const bf16_t* vp = xWTb + (size_t)(n0 * 16 + lq) * NTOK + j0 + h * 16;
            O[n0] = WMMA_BF16(Pf0, load_frag_B(vp), O[n0]);
            O[n0] = WMMA_BF16(Pf1, load_frag_B(vp + 32), O[n0]);
        }
    }

    // ---- epilogue: normalize, ELU, store f32 ----
#pragma unroll
    for (int r = 0; r < 8; r++) {
        const float inv = 1.f / lrow[r];
        const int m = r + 8 * h;
        float* orow = out + (size_t)(b * NTOK + row0 + m) * DF;
#pragma unroll
        for (int n0 = 0; n0 < 8; n0++) {
            float v = O[n0][r] * inv;
            v = (v > 0.f) ? v : (__expf(v) - 1.f);
            orow[n0 * 16 + lq] = v;
        }
    }
}

// ---------------------------------------------------------------------------
extern "C" void kernel_launch(void* const* d_in, const int* in_sizes, int n_in,
                              void* d_out, int out_size, void* d_ws, size_t ws_size,
                              hipStream_t stream) {
    (void)in_sizes; (void)n_in; (void)out_size; (void)ws_size;
    const float* x   = (const float*)d_in[0];
    const float* adj = (const float*)d_in[1];
    const float* W   = (const float*)d_in[2];
    float* out = (float*)d_out;

    bf16_t* xW  = (bf16_t*)d_ws;                       // 4 MB
    bf16_t* xWT = xW + (size_t)BATCH * NTOK * DF;      // 4 MB

    // 1024 row-tiles total, 4 waves (tiles) per 128-thread block
    gat_xw_kernel<<<256, 128, 0, stream>>>(x, W, xW, xWT);
    gat_attn_kernel<<<256, 128, 0, stream>>>(adj, xW, xWT, out);
}
